// IBC3d_70557722739463
// MI455X (gfx1250) — matrix operand order repaired and checked by reading, not compiled, not run on hardware
//
#include <hip/hip_runtime.h>

typedef __attribute__((ext_vector_type(16))) _Float16     v16h;
typedef __attribute__((ext_vector_type(8)))  float        v8f;
typedef __attribute__((ext_vector_type(4)))  unsigned int v4u;

#define B_   2
#define CIN  32
#define COUT 16
#define DD   48
#define HIN  64
#define WIN  128
#define HUP  128
#define WUP  256
#define NTAP 27
#define WSLOTS 258                      // 256 + 2 zero halo columns
#define ROW_HALVES (WSLOTS * CIN)       // 8256 halves per (n,d,h') row

// ---------------------------------------------------------------------------
// Prep: alpha = mean|W| per Cout; fold alpha into BN scale; pack sign(W) into
// the WMMA 16x16x32 f16 A operand in k-contiguous order: Apack[tap][lane][j], where
// half j of each lane's v16h maps to K = j + 16*(lane>=16),
// row M = lane & 15.   (CDNA5 ISA "16-bit A-Matrix 16x32")
// ---------------------------------------------------------------------------
__global__ void ibc3d_prep(const float* __restrict__ w,
                           const float* __restrict__ gamma,
                           const float* __restrict__ beta,
                           const float* __restrict__ rmean,
                           const float* __restrict__ rvar,
                           float* __restrict__ scaleA,
                           float* __restrict__ shiftA,
                           _Float16* __restrict__ Apack) {
    int tid = threadIdx.x;
    if (tid < COUT) {
        float s = 0.f;
        for (int i = 0; i < CIN * NTAP; ++i) s += fabsf(w[tid * CIN * NTAP + i]);
        float alpha = s / (float)(CIN * NTAP);
        float inv   = gamma[tid] * rsqrtf(rvar[tid] + 1e-5f);
        scaleA[tid] = alpha * inv;                  // conv out = alpha * binDot
        shiftA[tid] = beta[tid] - rmean[tid] * inv;
    }
    for (int idx = tid; idx < NTAP * 32 * 16; idx += blockDim.x) {
        int tap  = idx >> 9;
        int lane = (idx >> 4) & 31;
        int j    = idx & 15;
        int K = j + 16 * (lane >= 16 ? 1 : 0);  // ci
        int M = lane & 15;                                          // co
        float v = w[(M * CIN + K) * NTAP + tap];
        Apack[idx] = (v > 0.f) ? (_Float16)1.0f
                               : ((v < 0.f) ? (_Float16)-1.0f : (_Float16)0.0f);
    }
}

// ---------------------------------------------------------------------------
// Binarize: write sign(upsample(x)) ONCE per element (vs 9x halo redundancy)
// into xbin[n][d][h'][wslot(258)][ci(32)] f16, ci-contiguous (= B-operand
// per-lane layout), with zero halo at wslot 0 and 257.  Coalesced f32 reads
// (lanes along w), LDS transpose to [w][ci], 16B vector stores out.
// One block per (n, d, h') row.
// ---------------------------------------------------------------------------
__global__ void __launch_bounds__(256)
ibc3d_binarize(const float* __restrict__ x, _Float16* __restrict__ xbin) {
    __shared__ _Float16 srow[ROW_HALVES];           // 16.5 KB

    int bx = blockIdx.x;
    int hp = bx % HUP;
    int d  = (bx / HUP) % DD;
    int n  = bx / (HUP * DD);
    int tid = threadIdx.x;

    // h-direction half-pixel scale-2 weights (uniform per block)
    int ih = hp >> 1, hi0, hi1; float wh0, wh1;
    if ((hp & 1) == 0) { hi0 = (ih > 0) ? ih - 1 : 0; hi1 = ih; wh0 = 0.25f; wh1 = 0.75f; }
    else               { hi0 = ih; hi1 = (ih + 1 < HIN) ? ih + 1 : HIN - 1; wh0 = 0.75f; wh1 = 0.25f; }

    const float* xnd = x + ((size_t)n * CIN * DD + d) * (HIN * WIN);

    for (int idx = tid; idx < WUP * CIN; idx += 256) {
        int ci = idx >> 8;             // lanes vary along w -> coalesced reads
        int w  = idx & 255;
        int kw_ = w >> 1, wi0, wi1; float ww0, ww1;
        if ((w & 1) == 0) { wi0 = (kw_ > 0) ? kw_ - 1 : 0; wi1 = kw_; ww0 = 0.25f; ww1 = 0.75f; }
        else              { wi0 = kw_; wi1 = (kw_ + 1 < WIN) ? kw_ + 1 : WIN - 1; ww0 = 0.75f; ww1 = 0.25f; }

        const float* p = xnd + (size_t)ci * (DD * HIN * WIN);
        float r0 = ww0 * p[hi0 * WIN + wi0] + ww1 * p[hi0 * WIN + wi1];
        float r1 = ww0 * p[hi1 * WIN + wi0] + ww1 * p[hi1 * WIN + wi1];
        float v  = wh0 * r0 + wh1 * r1;
        srow[(w + 1) * CIN + ci] = (v > 0.f) ? (_Float16)1.0f
                                 : ((v < 0.f) ? (_Float16)-1.0f : (_Float16)0.0f);
    }
    if (tid < CIN)           srow[tid] = (_Float16)0.0f;                 // wslot 0
    else if (tid < 2 * CIN)  srow[257 * CIN + (tid - CIN)] = (_Float16)0.0f; // wslot 257
    __syncthreads();

    _Float16* dst = xbin + (size_t)(((n * DD + d) * HUP) + hp) * ROW_HALVES;
    const v4u* s4 = (const v4u*)srow;
    v4u*       d4 = (v4u*)dst;
    for (int c = tid; c < ROW_HALVES / 8; c += 256) d4[c] = s4[c];       // 16B stores
}

// ---------------------------------------------------------------------------
// Conv: one block (256 thr = 8 wave32) per (n, d, h') row; each wave owns
// 32 w' (two 16x16 tiles, A reused across both).  d/h zero-pad handled by
// skipping taps with wave-uniform scalar branches (EXEC stays all-1s).
// B operand: one 32B contiguous L2-resident read per lane per tile.
// ---------------------------------------------------------------------------
__global__ void __launch_bounds__(256)
ibc3d_conv(const _Float16* __restrict__ xbin,
           const float* __restrict__ scaleA,
           const float* __restrict__ shiftA,
           const _Float16* __restrict__ Apack,
           float* __restrict__ out) {
    int bx = blockIdx.x;
    int hp = bx % HUP;
    int d  = (bx / HUP) % DD;
    int n  = bx / (HUP * DD);
    int tid = threadIdx.x;

    int wave   = tid >> 5;
    int lane   = tid & 31;
    int ncol   = lane & 15;       // N within tile
    int hihalf = lane >> 4;       // B/C hi-half (ci 16..31 / M + 8)
    int wbase  = wave * 32;       // two tiles: [wbase, wbase+16)

    v8f c0 = {}, c1 = {};
    const v16h* Ap = (const v16h*)Apack;

#pragma unroll
    for (int kd = 0; kd < 3; ++kd) {
        int dg = d + kd - 1;
        if (dg < 0 || dg >= DD) continue;          // zero-pad plane: skip taps
#pragma unroll
        for (int kh = 0; kh < 3; ++kh) {
            int hg = hp + kh - 1;
            if (hg < 0 || hg >= HUP) continue;     // zero-pad row: skip taps
            const _Float16* row =
                xbin + (size_t)(((n * DD + dg) * HUP) + hg) * ROW_HALVES;
            if (hg + 1 < HUP)
                __builtin_prefetch(row + ROW_HALVES, 0, 1);  // next h' row
#pragma unroll
            for (int kw = 0; kw < 3; ++kw) {
                int tap = (kd * 3 + kh) * 3 + kw;
                v16h a = Ap[tap * 32 + lane];
                const _Float16* bp = row + (wbase + ncol + kw) * CIN + hihalf * 16;
                v16h b0 = *(const v16h*)bp;
                v16h b1 = *(const v16h*)(bp + 16 * CIN);
                c0 = __builtin_amdgcn_wmma_f32_16x16x32_f16(
                         false, a, false, b0, (short)0, c0, false, false);
                c1 = __builtin_amdgcn_wmma_f32_16x16x32_f16(
                         false, a, false, b1, (short)0, c1, false, false);
            }
        }
    }

    // epilogue: D layout -> VGPR r, lane-hi: M = r + 8*hihalf, N = lane&15
#pragma unroll
    for (int r = 0; r < 8; ++r) {
        int co = r + 8 * hihalf;
        float sA = scaleA[co], sh = shiftA[co];
        float v0 = c0[r] * sA + sh; v0 = v0 > 0.f ? v0 : 0.f;
        float v1 = c1[r] * sA + sh; v1 = v1 > 0.f ? v1 : 0.f;
        size_t base = ((((size_t)n * COUT + co) * DD + d) * HUP + hp) * (size_t)WUP;
        out[base + wbase + ncol]      = v0;
        out[base + wbase + 16 + ncol] = v1;
    }
}

// ---------------------------------------------------------------------------
extern "C" void kernel_launch(void* const* d_in, const int* in_sizes, int n_in,
                              void* d_out, int out_size, void* d_ws, size_t ws_size,
                              hipStream_t stream) {
    const float* x     = (const float*)d_in[0];
    const float* wgt   = (const float*)d_in[1];
    const float* gamma = (const float*)d_in[2];
    const float* beta  = (const float*)d_in[3];
    const float* rmean = (const float*)d_in[4];
    const float* rvar  = (const float*)d_in[5];

    // ws layout:
    //   [0,64)B      scaleA (16 f32)
    //   [64,128)B    shiftA (16 f32)
    //   [128,...)    Apack: 27*32*16 f16 = 27648 B
    //   [32768,...)  xbin:  2*48*128*258*32 f16 = 193.5 MiB
    float*    scaleA = (float*)d_ws;
    float*    shiftA = scaleA + 16;
    _Float16* Apack  = (_Float16*)((char*)d_ws + 128);
    _Float16* xbin   = (_Float16*)((char*)d_ws + 32768);

    ibc3d_prep<<<1, 64, 0, stream>>>(wgt, gamma, beta, rmean, rvar,
                                     scaleA, shiftA, Apack);

    int nRows = B_ * DD * HUP;                       // 12288
    ibc3d_binarize<<<nRows, 256, 0, stream>>>(x, xbin);
    ibc3d_conv<<<nRows, 256, 0, stream>>>(xbin, scaleA, shiftA, Apack,
                                          (float*)d_out);
}
